// SimpleMessageLayer_48567490183739
// MI455X (gfx1250) — matrix operand order repaired and checked by reading, not compiled, not run on hardware
//
#include <hip/hip_runtime.h>
#include <hip/hip_bf16.h>

#define N_NODES 50000
#define N_EDGES 800000
#define D_NODE  64
#define D_EDGE  32
#define MSG     16

typedef __attribute__((ext_vector_type(16))) __bf16 v16bf;
typedef __attribute__((ext_vector_type(8)))  float  v8f;
typedef __attribute__((ext_vector_type(4)))  float  fvec4;

// Pack 16 f32 -> v16bf in WMMA element order (elements 0..15).
__device__ __forceinline__ v16bf pack16(fvec4 a, fvec4 b, fvec4 c, fvec4 d) {
    v16bf r;
    r[0]=(__bf16)a[0];  r[1]=(__bf16)a[1];  r[2]=(__bf16)a[2];  r[3]=(__bf16)a[3];
    r[4]=(__bf16)b[0];  r[5]=(__bf16)b[1];  r[6]=(__bf16)b[2];  r[7]=(__bf16)b[3];
    r[8]=(__bf16)c[0];  r[9]=(__bf16)c[1];  r[10]=(__bf16)c[2]; r[11]=(__bf16)c[3];
    r[12]=(__bf16)d[0]; r[13]=(__bf16)d[1]; r[14]=(__bf16)d[2]; r[15]=(__bf16)d[3];
    return r;
}

__device__ __forceinline__ void atomAddF32(float* p, float v) {
    // Hardware global_atomic_add_f32 (relaxed, device scope).
    __hip_atomic_fetch_add(p, v, __ATOMIC_RELAXED, __HIP_MEMORY_SCOPE_AGENT);
}

__global__ void zero_kernel(float* __restrict__ p, int n) {
    int i = blockIdx.x * blockDim.x + threadIdx.x;
    if (i < n) p[i] = 0.0f;
}

// One wave == 16 edges. messages[16x16] = A(16x160) x B(160x16) via 5 bf16 WMMAs,
// then bias + leaky_relu + softmax(16) via 16-lane butterflies, then fp32 atomic
// scatter into sums[n1].
__global__ __launch_bounds__(256) void edge_msg_kernel(
    const float* __restrict__ nf, const int* __restrict__ n1,
    const int* __restrict__ n2, const float* __restrict__ ef,
    const float* __restrict__ fe_w, const float* __restrict__ fe_b,
    float* __restrict__ sums, int numTiles)
{
    int wave = blockIdx.x * 8 + (threadIdx.x >> 5);
    if (wave >= numTiles) return;          // uniform per wave -> EXEC stays all-ones
    int lane = threadIdx.x & 31;
    int grp  = lane >> 4;                  // K-half selector
    int ml   = lane & 15;                  // row (A) / column (B,C)
    int kb   = grp * 16;

    // B fragments: B[k][n] = fe_w[n][k]; lane n = ml, K = 32s + 16*grp + e
    v16bf bw[5];
#pragma unroll
    for (int s = 0; s < 5; ++s) {
        const float* p = fe_w + ml * 160 + s * 32 + grp * 16;
        bw[s] = pack16(*(const fvec4*)(p),     *(const fvec4*)(p + 4),
                       *(const fvec4*)(p + 8), *(const fvec4*)(p + 12));
    }

    int eBase = wave * 16;
    int e  = eBase + ml;                   // this lane gathers features for edge row ml
    int i1 = n1[e];
    int i2 = n2[e];

    v8f acc = {};
#pragma unroll
    for (int s = 0; s < 5; ++s) {
        const float* base =
            (s < 2) ? nf + (long)i1 * D_NODE + s * 32 :
            (s < 4) ? nf + (long)i2 * D_NODE + (s - 2) * 32 :
                      ef + (long)e  * D_EDGE;
        // A lane layout: K run {kb..kb+15} within this 32-K step
        fvec4 r0 = *(const fvec4*)(base + kb);
        fvec4 r1 = *(const fvec4*)(base + kb + 4);
        fvec4 r2 = *(const fvec4*)(base + kb + 8);
        fvec4 r3 = *(const fvec4*)(base + kb + 12);
        v16bf av = pack16(r0, r1, r2, r3);
        acc = __builtin_amdgcn_wmma_f32_16x16x32_bf16(false, av, false, bw[s],
                                                      (short)0, acc, false, false);
    }

    float bias = fe_b[ml];                 // bias indexed by output channel n = ml
#pragma unroll
    for (int r = 0; r < 8; ++r) {
        float v = acc[r] + bias;           // row M = r + 8*grp, col N = ml
        v = v > 0.0f ? v : 0.01f * v;      // leaky_relu(0.01)
        // softmax over the 16 channels: butterfly within each 16-lane half
        float mx = v;
#pragma unroll
        for (int off = 1; off <= 8; off <<= 1)
            mx = fmaxf(mx, __shfl_xor(mx, off, 32));
        float ex = __expf(v - mx);
        float sm = ex;
#pragma unroll
        for (int off = 1; off <= 8; off <<= 1)
            sm += __shfl_xor(sm, off, 32);
        float f = ex / sm;
        // scatter by n1 of edge (eBase + r + 8*grp); that index lives in lane r+8*grp
        int idx = __shfl(i1, r + 8 * grp, 32);
        atomAddF32(sums + (long)idx * MSG + ml, f);
    }
}

// One wave == 16 nodes. z[16x64] = A(16x80, zero-padded to 96) x fh_w^T,
// 4 column tiles x 3 K-steps of bf16 WMMA.
__global__ __launch_bounds__(256) void node_update_kernel(
    const float* __restrict__ nf, const float* __restrict__ sums,
    const float* __restrict__ fh_w, const float* __restrict__ fh_b,
    float* __restrict__ out, int numTiles)
{
    int wave = blockIdx.x * 8 + (threadIdx.x >> 5);
    if (wave >= numTiles) return;
    int lane = threadIdx.x & 31;
    int grp  = lane >> 4;
    int ml   = lane & 15;
    int kb   = grp * 16;

    int nBase = wave * 16;
    int node  = nBase + ml;

    // A fragments: k 0..63 = node_features, k 64..79 = sums, k 80..95 = 0
    v16bf av[3];
    {
        const float* base = nf + (long)node * D_NODE;
        av[0] = pack16(*(const fvec4*)(base + kb),      *(const fvec4*)(base + kb + 4),
                       *(const fvec4*)(base + kb + 8),  *(const fvec4*)(base + kb + 12));
        base += 32;
        av[1] = pack16(*(const fvec4*)(base + kb),      *(const fvec4*)(base + kb + 4),
                       *(const fvec4*)(base + kb + 8),  *(const fvec4*)(base + kb + 12));
        const float* sp = sums + (long)node * MSG;      // K run 64..79 (grp 0)
        fvec4 z  = {0.0f, 0.0f, 0.0f, 0.0f};            // K 80.. -> zero pad
        if (grp == 0)
            av[2] = pack16(*(const fvec4*)(sp), *(const fvec4*)(sp + 4),
                           *(const fvec4*)(sp + 8), *(const fvec4*)(sp + 12));
        else
            av[2] = pack16(z, z, z, z);
    }

#pragma unroll
    for (int nt = 0; nt < 4; ++nt) {
        v8f acc = {};
#pragma unroll
        for (int s = 0; s < 3; ++s) {
            v16bf bwf;
            int col = s * 32 + grp * 16;                // K offset into fh_w row
            if (col < 80) {
                const float* p = fh_w + (nt * 16 + ml) * 80 + col;
                bwf = pack16(*(const fvec4*)(p),     *(const fvec4*)(p + 4),
                             *(const fvec4*)(p + 8), *(const fvec4*)(p + 12));
            } else {                                    // zero-padded K 80..95
#pragma unroll
                for (int i = 0; i < 16; ++i) bwf[i] = (__bf16)0.0f;
            }
            acc = __builtin_amdgcn_wmma_f32_16x16x32_bf16(false, av[s], false, bwf,
                                                          (short)0, acc, false, false);
        }
        float bias = fh_b[nt * 16 + ml];                // per output channel n
#pragma unroll
        for (int r = 0; r < 8; ++r) {
            int m = r + 8 * grp;
            out[(long)(nBase + m) * D_NODE + nt * 16 + ml] = acc[r] + bias;
        }
    }
}

extern "C" void kernel_launch(void* const* d_in, const int* in_sizes, int n_in,
                              void* d_out, int out_size, void* d_ws, size_t ws_size,
                              hipStream_t stream) {
    const float* nf   = (const float*)d_in[0];
    const int*   eni  = (const int*)d_in[1];        // [2, N_EDGES] int32
    const float* ef   = (const float*)d_in[2];
    const float* fe_w = (const float*)d_in[3];
    const float* fe_b = (const float*)d_in[4];
    const float* fh_w = (const float*)d_in[5];
    const float* fh_b = (const float*)d_in[6];
    float* out  = (float*)d_out;
    float* sums = (float*)d_ws;                     // [N_NODES, MSG] f32 = 3.2 MB

    int sumN = N_NODES * MSG;
    zero_kernel<<<(sumN + 255) / 256, 256, 0, stream>>>(sums, sumN);

    int edgeTiles = N_EDGES / 16;                   // 50000 wave-tiles
    edge_msg_kernel<<<(edgeTiles + 7) / 8, 256, 0, stream>>>(
        nf, eni, eni + N_EDGES, ef, fe_w, fe_b, sums, edgeTiles);

    int nodeTiles = N_NODES / 16;                   // 3125 wave-tiles
    node_update_kernel<<<(nodeTiles + 7) / 8, 256, 0, stream>>>(
        nf, sums, fh_w, fh_b, out, nodeTiles);
}
